// Encoder_26749056319675
// MI455X (gfx1250) — matrix operand order repaired and checked by reading, not compiled, not run on hardware
//
#include <hip/hip_runtime.h>
#include <hip/hip_bf16.h>
#include <math.h>

// ---------------------------------------------------------------------------
// Transformer encoder for MI455X (gfx1250): bf16 WMMA everywhere, flash attn,
// TDM (tensor_load_to_lds) double-buffered weight staging in the GEMMs.
// ---------------------------------------------------------------------------

typedef __bf16 bf16;
typedef __attribute__((ext_vector_type(16))) __bf16 v16bf;
typedef __attribute__((ext_vector_type(8)))  __bf16 v8bf;
typedef __attribute__((ext_vector_type(8)))  float  v8f;
typedef __attribute__((ext_vector_type(4)))  unsigned int u32x4;
typedef __attribute__((ext_vector_type(4)))  int i32x4;
typedef __attribute__((ext_vector_type(8)))  int i32x8;

#define LAYERS 6
#define DIM    1024
#define SEQ    2048
#define FF     4096
#define HEADS  16
#define HDIM   64
#define BATCH  2
#define MROWS  (BATCH * SEQ)   // 4096

#if __has_builtin(__builtin_amdgcn_tensor_load_to_lds)
#define HAVE_TDM 1
#else
#define HAVE_TDM 0
#endif

// ---------------- WMMA helpers --------------------------------------------

__device__ inline v8f wmma_bf16(v16bf a, v16bf b, v8f c) {
  // D = A(16x32 bf16) * B(32x16 bf16) + C(16x16 f32)
  return __builtin_amdgcn_wmma_f32_16x16x32_bf16(false, a, false, b, (short)0, c,
                                                 false, false);
}

__device__ inline v8f zero8() {
  v8f z;
#pragma unroll
  for (int i = 0; i < 8; ++i) z[i] = 0.0f;
  return z;
}

// A fragment (16x32, row-major source, lda in elements).
// ISA layout: lane l(lo): VGPR0-3 = K[k0..k0+7], VGPR4-7 = K[k0+16..k0+23];
// hi half is +8 in K.
__device__ inline v16bf load_frag_a(const bf16* A, int lda, int row0, int k0, int lane) {
  int l = lane & 15, h = lane >> 4;
  const bf16* p = A + (size_t)(row0 + l) * lda + k0 + h * 8;
  v8bf lo = *(const v8bf*)(p);
  v8bf hi = *(const v8bf*)(p + 16);
  v16bf r;
#pragma unroll
  for (int i = 0; i < 8; ++i) { r[i] = lo[i]; r[i + 8] = hi[i]; }
  return r;
}

// B fragment (32x16) from N-major storage Bt[N][K] (ldb in elements).
// ISA layout: lane l(lo): VGPR0-3 = K[k0..k0+7], VGPR4-7 = K[k0+16..k0+23]; hi half is +8 in K.
__device__ inline v16bf load_frag_b(const bf16* Bt, int ldb, int n0, int k0, int lane) {
  int n = lane & 15, h = lane >> 4;
  const bf16* p = Bt + (size_t)(n0 + n) * ldb + k0 + h * 8;
  v8bf lo = *(const v8bf*)(p);
  v8bf hi = *(const v8bf*)(p + 16);
  v16bf r;
#pragma unroll
  for (int i = 0; i < 8; ++i) { r[i] = lo[i]; r[i + 8] = hi[i]; }
  return r;
}

// ---------------- TDM: DMA a 128x32-element bf16 tile into LDS ------------
// D# per CDNA5 ISA: group0 = {flags, lds_addr, global_addr,
// type=2}; group1 = {data_size, tensor dims (huge, no clip), tile dims
// 32x128, dim0 stride = K}. Groups 2/3 zero (2D). TENSORcnt tracks completion.

__device__ inline void tdm_stage_tile(bf16* lds, const bf16* gsrc, int strideK, int lane) {
#if HAVE_TDM
  unsigned long long ga = (unsigned long long)(size_t)gsrc;
  unsigned ldsOff = (unsigned)(size_t)lds;  // flat->LDS: low 32 bits are the LDS offset
  u32x4 g0;
  g0[0] = 1u;                                   // count=1, user descriptor
  g0[1] = ldsOff;                               // lds_addr (bytes)
  g0[2] = (unsigned)ga;                         // global_addr[31:0]
  g0[3] = (unsigned)((ga >> 32) & 0x01FFFFFFu) | (2u << 30);  // addr[56:32] | type=2
  i32x8 g1;
  g1[0] = 0x00010000;                           // workgroup_mask=0, data_size=1 (2 bytes)
  g1[1] = (int)0xFFFF0000u;                     // tensor_dim0[15:0]=0xFFFF (huge)
  g1[2] = (int)0xFFFF7FFFu;                     // dim0[31:16]=0x7FFF, dim1[15:0]=0xFFFF
  g1[3] = (int)((32u << 16) | 0x7FFFu);         // dim1[31:16]=0x7FFF, tile_dim0=32
  g1[4] = 128;                                  // tile_dim1=128, tile_dim2=0
  g1[5] = strideK;                              // tensor_dim0_stride[31:0] (elements)
  g1[6] = 0;                                    // stride0 hi, dim1_stride lo
  g1[7] = 0;
  i32x4 gz;
  gz[0] = 0; gz[1] = 0; gz[2] = 0; gz[3] = 0;
#if defined(__clang_major__) && (__clang_major__ >= 23)
  i32x8 gz8;
#pragma unroll
  for (int i = 0; i < 8; ++i) gz8[i] = 0;
  __builtin_amdgcn_tensor_load_to_lds(g0, g1, gz, gz, gz8, 0);
#else
  __builtin_amdgcn_tensor_load_to_lds(g0, g1, gz, gz, 0);
#endif
  (void)lane;
#else
  // Fallback: cooperative copy by the calling wave (32 lanes, 4 rows each).
  for (int r = lane; r < 128; r += 32) {
    v8bf* d = (v8bf*)(lds + r * 32);
    const v8bf* s = (const v8bf*)(gsrc + (size_t)r * strideK);
    d[0] = s[0]; d[1] = s[1]; d[2] = s[2]; d[3] = s[3];
  }
#endif
}

__device__ inline void tdm_fence() {
#if HAVE_TDM
  __builtin_amdgcn_s_wait_tensorcnt(0);
#endif
}

// ---------------- DPP 16-lane butterfly reductions (pure VALU) ------------

template <int CTRL>
__device__ inline float dpp_xf(float x) {
  int i = __float_as_int(x);
  int r = __builtin_amdgcn_update_dpp(i, i, CTRL, 0xf, 0xf, true);
  return __int_as_float(r);
}

__device__ inline float red16_max(float x) {
  x = fmaxf(x, dpp_xf<0xB1>(x));   // quad_perm [1,0,3,2]  (xor 1)
  x = fmaxf(x, dpp_xf<0x4E>(x));   // quad_perm [2,3,0,1]  (xor 2)
  x = fmaxf(x, dpp_xf<0x141>(x));  // row_half_mirror      (xor 7 ~ xor 4)
  x = fmaxf(x, dpp_xf<0x140>(x));  // row_mirror           (xor 15 ~ xor 8)
  return x;
}

__device__ inline float red16_sum(float x) {
  x += dpp_xf<0xB1>(x);
  x += dpp_xf<0x4E>(x);
  x += dpp_xf<0x141>(x);
  x += dpp_xf<0x140>(x);
  return x;
}

// ---------------- Weight transpose f32[R][C] -> bf16[C][R] ----------------

__global__ void transpose_to_bf16(const float* __restrict__ in, bf16* __restrict__ out,
                                  int R, int C) {
  __shared__ float tile[32][33];
  int c = blockIdx.x * 32 + threadIdx.x;
  int r = blockIdx.y * 32 + threadIdx.y;
#pragma unroll
  for (int i = 0; i < 32; i += 8)
    if (r + i < R && c < C) tile[threadIdx.y + i][threadIdx.x] = in[(size_t)(r + i) * C + c];
  __syncthreads();
  int oc = blockIdx.y * 32 + threadIdx.x;
  int orow = blockIdx.x * 32 + threadIdx.y;
#pragma unroll
  for (int i = 0; i < 32; i += 8)
    if (orow + i < C && oc < R)
      out[(size_t)(orow + i) * R + oc] = (bf16)tile[threadIdx.x][threadIdx.y + i];
}

// ---------------- Embedding + sinusoidal positional encoding --------------

__global__ void embed_kernel(const int* __restrict__ tokens, const float* __restrict__ emb,
                             float* __restrict__ xf, bf16* __restrict__ xb) {
  int row = blockIdx.x;                // b*SEQ + s
  int s   = row & (SEQ - 1);
  int tok = tokens[row];
  const float kLogBase = 9.210340371976184f / (float)DIM;  // ln(10000)/D
#pragma unroll
  for (int j = 0; j < 4; ++j) {
    int d = threadIdx.x + j * 256;
    int i = d >> 1;
    float f   = __expf(-(float)(2 * i) * kLogBase);
    float arg = (float)s * f;
    float pe  = (d & 1) ? __cosf(arg) : __sinf(arg);
    float v   = emb[(size_t)tok * DIM + d] * 32.0f + pe;  // sqrt(1024) = 32
    xf[(size_t)row * DIM + d] = v;
    xb[(size_t)row * DIM + d] = (bf16)v;
  }
}

// ---------------- GEMM: C[M][N] = A[M][K] * Bt[N][K]^T + bias -------------
// 256 threads = 8 waves; wave tile 32x64; block 128x128. B tile (128x32)
// staged into LDS by the TDM, double-buffered, shared by all 8 waves.

template <bool RELU, bool OUT_BF16>
__global__ __launch_bounds__(256, 1) void gemm_bf16(
    const bf16* __restrict__ A, const bf16* __restrict__ Bt,
    const float* __restrict__ bias, void* __restrict__ out, int N, int K) {
  __shared__ __align__(16) bf16 bTile[2][128 * 32];

  int lane = threadIdx.x & 31;
  int wave = threadIdx.x >> 5;
  int wm = wave & 3, wn = wave >> 2;
  int row0 = blockIdx.y * 128 + wm * 32;
  int nblk = blockIdx.x * 128;
  int n0   = wn * 64;                 // N offset inside the LDS tile

  v8f acc[2][4];
#pragma unroll
  for (int mi = 0; mi < 2; ++mi)
#pragma unroll
    for (int t = 0; t < 4; ++t) acc[mi][t] = zero8();

  const bf16* bBase = Bt + (size_t)nblk * K;
  if (wave == 0) tdm_stage_tile(&bTile[0][0], bBase, K, lane);

  for (int k0 = 0; k0 < K; k0 += 32) {
    int buf = (k0 >> 5) & 1;
    tdm_fence();        // wave0: DMA for bTile[buf] complete (no-op for others)
    __syncthreads();    // tile visible to all waves; prior reads of buf^1 done
    if (k0 + 32 < K && wave == 0)
      tdm_stage_tile(&bTile[buf ^ 1][0], bBase + (k0 + 32), K, lane);

    v16bf a0 = load_frag_a(A, K, row0, k0, lane);
    v16bf a1 = load_frag_a(A, K, row0 + 16, k0, lane);
#pragma unroll
    for (int t = 0; t < 4; ++t) {
      v16bf bfr = load_frag_b(&bTile[buf][0], 32, n0 + t * 16, 0, lane);
      acc[0][t] = wmma_bf16(a0, bfr, acc[0][t]);
      acc[1][t] = wmma_bf16(a1, bfr, acc[1][t]);
    }
    if (k0 + 32 < K)   // gfx1250 global_prefetch_b8 path for the A stream
      __builtin_prefetch(A + (size_t)(row0 + (lane & 15)) * K + k0 + 32, 0, 1);
  }

  int hh  = lane >> 4;
  int col = lane & 15;
  float bv[4];
#pragma unroll
  for (int t = 0; t < 4; ++t) bv[t] = bias[nblk + n0 + t * 16 + col];
#pragma unroll
  for (int mi = 0; mi < 2; ++mi) {
#pragma unroll
    for (int r = 0; r < 8; ++r) {
      int row = row0 + mi * 16 + r + hh * 8;
#pragma unroll
      for (int t = 0; t < 4; ++t) {
        float v = acc[mi][t][r] + bv[t];
        if (RELU) v = fmaxf(v, 0.0f);
        size_t idx = (size_t)row * N + nblk + n0 + t * 16 + col;
        if (OUT_BF16) ((bf16*)out)[idx] = (bf16)v;
        else          ((float*)out)[idx] = v;
      }
    }
  }
}

// ---------------- Fused flash attention (one wave per 16-query tile) ------
// grid = (BATCH*HEADS, SEQ/64), block = 128 (4 waves, independent q tiles).

#define PLD 40   // padded LDS row stride (elements) to dodge bank conflicts

__global__ __launch_bounds__(128, 1) void attn_kernel(
    const bf16* __restrict__ Qb, const bf16* __restrict__ Kb,
    const bf16* __restrict__ Vb, bf16* __restrict__ Ob) {
  int lane = threadIdx.x & 31;
  int wave = threadIdx.x >> 5;
  int hh   = lane >> 4;
  int col  = lane & 15;

  int bh = blockIdx.x;
  int b  = bh / HEADS;
  int h  = bh - b * HEADS;
  int q0 = blockIdx.y * 64 + wave * 16;

  const bf16* Qh = Qb + (size_t)b * SEQ * DIM + h * HDIM;
  const bf16* Kh = Kb + (size_t)b * SEQ * DIM + h * HDIM;
  const bf16* Vh = Vb + (size_t)b * SEQ * DIM + h * HDIM;
  bf16*       Oh = Ob + (size_t)b * SEQ * DIM + h * HDIM;

  __shared__ bf16 smem[4 * (16 * PLD + HDIM * PLD)];
  bf16* pL = smem + wave * (16 * PLD + HDIM * PLD);
  bf16* vT = pL + 16 * PLD;

  v16bf aQ0 = load_frag_a(Qh, DIM, q0, 0, lane);
  v16bf aQ1 = load_frag_a(Qh, DIM, q0, 32, lane);

  float m[8], l[8];
  v8f acc[4];
#pragma unroll
  for (int r = 0; r < 8; ++r) { m[r] = -INFINITY; l[r] = 0.0f; }
#pragma unroll
  for (int t = 0; t < 4; ++t) acc[t] = zero8();

  for (int kt = 0; kt < SEQ; kt += 32) {
    // scores S[16q x 32k] = Q (16x64) * K^T, scaled by 1/sqrt(64)
    v8f s0 = wmma_bf16(aQ0, load_frag_b(Kh, DIM, kt, 0, lane), zero8());
    s0     = wmma_bf16(aQ1, load_frag_b(Kh, DIM, kt, 32, lane), s0);
    v8f s1 = wmma_bf16(aQ0, load_frag_b(Kh, DIM, kt + 16, 0, lane), zero8());
    s1     = wmma_bf16(aQ1, load_frag_b(Kh, DIM, kt + 16, 32, lane), s1);

#pragma unroll
    for (int r = 0; r < 8; ++r) { s0[r] *= 0.125f; s1[r] *= 0.125f; }

    // online softmax (row = r + 8*hh lives across the 16 lanes of a half)
#pragma unroll
    for (int r = 0; r < 8; ++r) {
      float tm = red16_max(fmaxf(s0[r], s1[r]));
      float mn = fmaxf(m[r], tm);
      float sc = __expf(m[r] - mn);
      m[r] = mn;
      float p0 = __expf(s0[r] - mn);
      float p1 = __expf(s1[r] - mn);
      float rs = red16_sum(p0 + p1);
      l[r] = l[r] * sc + rs;
#pragma unroll
      for (int t = 0; t < 4; ++t) acc[t][r] *= sc;
      s0[r] = p0;
      s1[r] = p1;
    }

    // P (C layout) -> LDS row-major [16][32] for A-layout reload
#pragma unroll
    for (int r = 0; r < 8; ++r) {
      int row = r + hh * 8;
      pL[row * PLD + col]      = (bf16)s0[r];
      pL[row * PLD + 16 + col] = (bf16)s1[r];
    }
    // V tile (32 keys x 64 d) -> LDS transposed vT[d][key]
    {
      const bf16* vrow = Vh + (size_t)(kt + lane) * DIM;
#pragma unroll
      for (int dd = 0; dd < HDIM; dd += 8) {
        v8bf vv = *(const v8bf*)(vrow + dd);
#pragma unroll
        for (int j = 0; j < 8; ++j) vT[(dd + j) * PLD + lane] = vv[j];
      }
    }
    __syncthreads();

    v16bf aP = load_frag_a(pL, PLD, 0, 0, lane);
#pragma unroll
    for (int t = 0; t < 4; ++t)
      acc[t] = wmma_bf16(aP, load_frag_b(vT, PLD, t * 16, 0, lane), acc[t]);
    __syncthreads();
  }

  // normalize and store (bf16, feeds O-projection GEMM)
#pragma unroll
  for (int r = 0; r < 8; ++r) {
    float inv = 1.0f / l[r];
    int row = q0 + r + hh * 8;
#pragma unroll
    for (int t = 0; t < 4; ++t)
      Oh[(size_t)row * DIM + t * 16 + col] = (bf16)(acc[t][r] * inv);
  }
}

// ---------------- Fused residual add + LayerNorm --------------------------

__global__ void add_ln_kernel(const float* x, const float* dlt,
                              const float* __restrict__ g, const float* __restrict__ be,
                              float* xo, bf16* xbo) {
  int row = blockIdx.x;
  int tid = threadIdx.x;
  const float* xr = x + (size_t)row * DIM;
  const float* dr = dlt + (size_t)row * DIM;
  float v[4];
  float s = 0.0f, q = 0.0f;
#pragma unroll
  for (int j = 0; j < 4; ++j) {
    int d = tid + j * 256;
    float t = xr[d] + dr[d];
    v[j] = t;
    s += t;
    q += t * t;
  }
#pragma unroll
  for (int msk = 16; msk >= 1; msk >>= 1) {
    s += __shfl_xor(s, msk, 32);
    q += __shfl_xor(q, msk, 32);
  }
  __shared__ float red[16];
  int w = tid >> 5;
  if ((tid & 31) == 0) { red[w] = s; red[8 + w] = q; }
  __syncthreads();
  if (tid == 0) {
    float ts = 0.0f, tq = 0.0f;
#pragma unroll
    for (int i = 0; i < 8; ++i) { ts += red[i]; tq += red[8 + i]; }
    red[0] = ts;
    red[8] = tq;
  }
  __syncthreads();
  float mean = red[0] * (1.0f / DIM);
  float var  = red[8] * (1.0f / DIM) - mean * mean;
  float rstd = rsqrtf(var + 1e-5f);
  float* xout = xo + (size_t)row * DIM;
  bf16*  xbr  = xbo + (size_t)row * DIM;
#pragma unroll
  for (int j = 0; j < 4; ++j) {
    int d = tid + j * 256;
    float y = (v[j] - mean) * rstd * g[d] + be[d];
    xout[d] = y;
    xbr[d]  = (bf16)y;
  }
}

// ---------------- Host-side orchestration ---------------------------------

extern "C" void kernel_launch(void* const* d_in, const int* in_sizes, int n_in,
                              void* d_out, int out_size, void* d_ws, size_t ws_size,
                              hipStream_t stream) {
  (void)in_sizes; (void)n_in; (void)out_size; (void)ws_size;

  const int*   tokens = (const int*)d_in[0];
  const float* emb    = (const float*)d_in[1];
  const float* Wq = (const float*)d_in[2];  const float* bq = (const float*)d_in[3];
  const float* Wk = (const float*)d_in[4];  const float* bk = (const float*)d_in[5];
  const float* Wv = (const float*)d_in[6];  const float* bv = (const float*)d_in[7];
  const float* Wo = (const float*)d_in[8];  const float* bo = (const float*)d_in[9];
  const float* W1 = (const float*)d_in[10]; const float* b1 = (const float*)d_in[11];
  const float* W2 = (const float*)d_in[12]; const float* b2 = (const float*)d_in[13];
  const float* g1 = (const float*)d_in[14]; const float* be1 = (const float*)d_in[15];
  const float* g2 = (const float*)d_in[16]; const float* be2 = (const float*)d_in[17];

  char* ws = (char*)d_ws;
  size_t off = 0;
  auto take = [&](size_t bytes) -> char* {
    char* p = ws + off;
    off += (bytes + 255) & ~(size_t)255;
    return p;
  };

  float* xf  = (float*)take((size_t)MROWS * DIM * 4);
  bf16*  xb  = (bf16*)take((size_t)MROWS * DIM * 2);
  bf16*  qb  = (bf16*)take((size_t)MROWS * DIM * 2);
  bf16*  kb  = (bf16*)take((size_t)MROWS * DIM * 2);
  bf16*  vb  = (bf16*)take((size_t)MROWS * DIM * 2);
  bf16*  ab  = (bf16*)take((size_t)MROWS * DIM * 2);
  float* of  = (float*)take((size_t)MROWS * DIM * 4);
  bf16*  h1b = (bf16*)take((size_t)MROWS * FF * 2);
  float* h2f = (float*)take((size_t)MROWS * DIM * 4);

  const size_t DD  = (size_t)DIM * DIM;
  const size_t DF  = (size_t)DIM * FF;
  const size_t WPL = 4 * DD + 2 * DF;  // bf16 elements per layer
  bf16* wT = (bf16*)take(LAYERS * WPL * 2);

  dim3 tb(32, 8);
  // 1) transpose all weights to bf16 [N][K]
  for (int lyr = 0; lyr < LAYERS; ++lyr) {
    bf16* base = wT + (size_t)lyr * WPL;
    transpose_to_bf16<<<dim3(DIM / 32, DIM / 32), tb, 0, stream>>>(Wq + lyr * DD, base + 0 * DD, DIM, DIM);
    transpose_to_bf16<<<dim3(DIM / 32, DIM / 32), tb, 0, stream>>>(Wk + lyr * DD, base + 1 * DD, DIM, DIM);
    transpose_to_bf16<<<dim3(DIM / 32, DIM / 32), tb, 0, stream>>>(Wv + lyr * DD, base + 2 * DD, DIM, DIM);
    transpose_to_bf16<<<dim3(DIM / 32, DIM / 32), tb, 0, stream>>>(Wo + lyr * DD, base + 3 * DD, DIM, DIM);
    transpose_to_bf16<<<dim3(FF / 32, DIM / 32), tb, 0, stream>>>(W1 + lyr * DF, base + 4 * DD, DIM, FF);
    transpose_to_bf16<<<dim3(DIM / 32, FF / 32), tb, 0, stream>>>(W2 + lyr * DF, base + 4 * DD + DF, FF, DIM);
  }

  // 2) embedding + positional encoding
  embed_kernel<<<MROWS, 256, 0, stream>>>(tokens, emb, xf, xb);

  // 3) layers
  dim3 gD(DIM / 128, MROWS / 128);
  dim3 gF(FF / 128, MROWS / 128);
  for (int lyr = 0; lyr < LAYERS; ++lyr) {
    bf16* wqT = wT + (size_t)lyr * WPL;
    bf16* wkT = wqT + DD;
    bf16* wvT = wkT + DD;
    bf16* woT = wvT + DD;
    bf16* w1T = woT + DD;
    bf16* w2T = w1T + DF;

    gemm_bf16<false, true><<<gD, 256, 0, stream>>>(xb, wqT, bq + lyr * DIM, qb, DIM, DIM);
    gemm_bf16<false, true><<<gD, 256, 0, stream>>>(xb, wkT, bk + lyr * DIM, kb, DIM, DIM);
    gemm_bf16<false, true><<<gD, 256, 0, stream>>>(xb, wvT, bv + lyr * DIM, vb, DIM, DIM);

    attn_kernel<<<dim3(BATCH * HEADS, SEQ / 64), 128, 0, stream>>>(qb, kb, vb, ab);

    gemm_bf16<false, false><<<gD, 256, 0, stream>>>(ab, woT, bo + lyr * DIM, of, DIM, DIM);
    add_ln_kernel<<<MROWS, 256, 0, stream>>>(xf, of, g1 + lyr * DIM, be1 + lyr * DIM, xf, xb);

    gemm_bf16<true, true><<<gF, 256, 0, stream>>>(xb, w1T, b1 + lyr * FF, h1b, FF, DIM);
    gemm_bf16<false, false><<<gD, 256, 0, stream>>>(h1b, w2T, b2 + lyr * DIM, h2f, DIM, FF);
    add_ln_kernel<<<MROWS, 256, 0, stream>>>(xf, h2f, g2 + lyr * DIM, be2 + lyr * DIM, xf, xb);
  }

  // 4) final residual -> output (f32)
  hipMemcpyAsync(d_out, xf, (size_t)MROWS * DIM * sizeof(float),
                 hipMemcpyDeviceToDevice, stream);
}
